// CrossAttention_72928544686517
// MI455X (gfx1250) — compile-verified, matrix operand order checked by reading, not run on hardware
//
#include <hip/hip_runtime.h>

#define B_   2
#define SQ_  2048
#define SKV_ 2048
#define D_   1024
#define H_   16
#define HD_  64

#define TK_   64   // K-tile staged per TDM load
#define LDA_  72   // padded LDS row stride in bf16 elems (64 data + 8 pad = 144B)

typedef __attribute__((ext_vector_type(16))) __bf16 bf16x16;
typedef __attribute__((ext_vector_type(8)))  float  f32x8;
typedef __attribute__((ext_vector_type(4)))  unsigned int u32x4;
typedef __attribute__((ext_vector_type(8)))  int  i32x8;
typedef __attribute__((ext_vector_type(4)))  int  i32x4;

__device__ __forceinline__ unsigned short f2bf(float f) {
  unsigned int u = __float_as_uint(f);
  if ((u & 0x7fffffffu) > 0x7f800000u) return (unsigned short)((u >> 16) | 0x0040u); // NaN
  u += 0x7fffu + ((u >> 16) & 1u); // round-to-nearest-even
  return (unsigned short)(u >> 16);
}

union FragU { uint4 u[2]; bf16x16 v; };

// Load a 16x32 bf16 WMMA fragment (A layout, == B layout applied to B^T stored [N][K]).
//   lanes 0-15 : row = lane,    K = k0+{0..7} and k0+{16..23}
//   lanes 16-31: row = lane-16, K = k0+{8..15} and k0+{24..31}
__device__ __forceinline__ bf16x16 load_frag(const unsigned short* base, int row0,
                                             int ld, int k0, int lane) {
  int r  = row0 + (lane & 15);
  int kb = k0 + ((lane >> 4) << 3);
  const unsigned short* p = base + (size_t)r * ld + kb;
  FragU f;
  f.u[0] = *(const uint4*)(p);
  f.u[1] = *(const uint4*)(p + 16);
  return f.v;
}

// ---- Tensor Data Mover: DMA a [128 rows x TK_ cols] bf16 tile from a row-major
// [rows][cols] tensor into LDS at lds_byte_off, padding +4 DWORDs every 32 DWORDs
// so each LDS row lands at stride LDA_ (=144B), killing ds_load_b128 bank conflicts.
__device__ __forceinline__ void tdm_load_tile(unsigned lds_byte_off,
                                              const unsigned short* gaddr,
                                              unsigned tensor_rows,
                                              unsigned tensor_cols) {
  unsigned long long ga = (unsigned long long)(uintptr_t)gaddr;
  u32x4 g0;
  g0[0] = 1u;                                   // count=1, user descriptor
  g0[1] = lds_byte_off;                         // lds_addr
  g0[2] = (unsigned)(ga & 0xffffffffu);         // global_addr[31:0]
  g0[3] = (unsigned)((ga >> 32) & 0x1ffffffu)   // global_addr[56:32]
        | (2u << 30);                           // type = 2 ("image")
  i32x8 g1;
  g1[0] = (1 << 16)    // data_size = 2 bytes
        | (1 << 20)    // pad_enable
        | (4 << 22)    // pad_interval: 32 DWORDs (code 4)
        | (3 << 25);   // pad_amount: 4 DWORDs (code 3)
  g1[1] = (int)(tensor_cols << 16);                                 // dim0[15:0]
  g1[2] = (int)((tensor_cols >> 16) & 0xffffu) | (int)(tensor_rows << 16); // dim0 hi | dim1 lo
  g1[3] = (int)((tensor_rows >> 16) & 0xffffu) | (TK_ << 16);       // dim1 hi | tile_dim0
  g1[4] = 128;                  // tile_dim1 = 128 rows, tile_dim2 = 0
  g1[5] = (int)tensor_cols;     // tensor_dim0_stride[31:0]
  g1[6] = 0;                    // stride hi, dim1_stride lo
  g1[7] = 0;
  i32x4 z4 = {0, 0, 0, 0};                      // 2-D tensor: groups 2/3 unused
  i32x8 z8 = {0, 0, 0, 0, 0, 0, 0, 0};
  __builtin_amdgcn_tensor_load_to_lds(g0, g1, z4, z4, z8, 0);
}

// ---------------- prep kernels ----------------

__global__ void cvt_bf16_kernel(const float* __restrict__ in,
                                unsigned short* __restrict__ out, int n) {
  int i = blockIdx.x * blockDim.x + threadIdx.x;
  int stride = gridDim.x * blockDim.x;
  for (; i < n; i += stride) out[i] = f2bf(in[i]);
}

// out[n*K + k] = bf16(in[k*N + n])   (W [K][N] -> W^T [N][K])
__global__ void transpose_cvt_kernel(const float* __restrict__ in,
                                     unsigned short* __restrict__ out, int K, int N) {
  int i = blockIdx.x * blockDim.x + threadIdx.x;
  int total = K * N;
  int stride = gridDim.x * blockDim.x;
  for (; i < total; i += stride) {
    int n = i / K, k = i - n * K;
    out[i] = f2bf(in[k * N + n]);
  }
}

// ---------------- TDM-fed bf16 WMMA GEMM ----------------
// C[M,N] = A[M,K] @ Bt[N,K]^T + bias[N]
// mode 0: write Q layout [B,H,SQ,HD] bf16 into out0
// mode 1: write K layout [B,H,SKV,HD] (n<D) into out0, V^T [B,H,HD,SKV] (n>=D) into out1
// mode 2: write f32 row-major [M,N] into outF
__global__ __launch_bounds__(256) void gemm_bf16_kernel(
    const unsigned short* __restrict__ A,
    const unsigned short* __restrict__ Bt,
    const float* __restrict__ bias,
    int M, int N, int K, int mode,
    unsigned short* __restrict__ out0,
    unsigned short* __restrict__ out1,
    float* __restrict__ outF) {
  // single shared object -> LDS offset 0; [buf0 A][buf0 B][buf1 A][buf1 B]
  __shared__ __align__(16) unsigned short smem[4 * 128 * LDA_];
  const unsigned TILE_B = 128u * LDA_ * 2u;   // bytes per 128xTK_ padded tile

  const int tid  = threadIdx.x;
  const int lane = tid & 31;
  const int w    = tid >> 5;
  const int wm   = w >> 1;   // 0..3
  const int wn   = w & 1;    // 0..1
  const int bm   = blockIdx.y * 128;
  const int bn   = blockIdx.x * 128;

  // Never true at runtime (M is 2048/4096 here), but the compiler cannot prove
  // it: keeps a visible store to smem so the optimizer cannot undef-fold the
  // fragment loads that consume data written by the (implicit) TDM DMA.
  if (M == 0x7fffffff) smem[tid] = 0;

  const unsigned short* Abase = A  + (size_t)bm * K;
  const unsigned short* Bbase = Bt + (size_t)bn * K;
  const int nk = K / TK_;

  f32x8 acc[2][4] = {};

  if (w == 0) {  // wave 0 drives the Tensor Data Mover
    tdm_load_tile(0 * TILE_B, Abase, (unsigned)M, (unsigned)K);
    tdm_load_tile(1 * TILE_B, Bbase, (unsigned)N, (unsigned)K);
  }

  for (int it = 0; it < nk; ++it) {
    const int cur = it & 1;
    if (w == 0) {
      if (it + 1 < nk) {
        const unsigned nxt = (unsigned)((it + 1) & 1);
        tdm_load_tile((2u * nxt + 0u) * TILE_B, Abase + (it + 1) * TK_,
                      (unsigned)M, (unsigned)K);
        tdm_load_tile((2u * nxt + 1u) * TILE_B, Bbase + (it + 1) * TK_,
                      (unsigned)N, (unsigned)K);
        __builtin_amdgcn_s_wait_tensorcnt(2);  // current tile done, next in flight
      } else {
        __builtin_amdgcn_s_wait_tensorcnt(0);
      }
    }
    __syncthreads();                     // tile `it` resident for all waves
    asm volatile("" ::: "memory");       // DMA-written LDS: no hoisting loads above

    const unsigned short* Asb = smem + (size_t)(2 * cur + 0) * (128 * LDA_);
    const unsigned short* Bsb = smem + (size_t)(2 * cur + 1) * (128 * LDA_);

#pragma unroll
    for (int kk = 0; kk < 2; ++kk) {       // two 32-deep WMMA sub-steps per tile
      bf16x16 af[2], bfr[4];
#pragma unroll
      for (int i = 0; i < 2; ++i) af[i]  = load_frag(Asb, wm * 32 + i * 16, LDA_, kk * 32, lane);
#pragma unroll
      for (int j = 0; j < 4; ++j) bfr[j] = load_frag(Bsb, wn * 64 + j * 16, LDA_, kk * 32, lane);
#pragma unroll
      for (int i = 0; i < 2; ++i)
#pragma unroll
        for (int j = 0; j < 4; ++j)
          acc[i][j] = __builtin_amdgcn_wmma_f32_16x16x32_bf16(
              false, af[i], false, bfr[j], (short)0, acc[i][j], false, false);
    }
    __syncthreads();  // all waves done with tile `it` before its buffer is re-DMAed
  }

  // epilogue: C layout -> lane l, element r is (m = r + 8*(l>=16), n = l&15)
#pragma unroll
  for (int i = 0; i < 2; ++i) {
#pragma unroll
    for (int j = 0; j < 4; ++j) {
      int gn = bn + wn * 64 + j * 16 + (lane & 15);
      float bv = bias[gn];
#pragma unroll
      for (int r = 0; r < 8; ++r) {
        int gm = bm + wm * 32 + i * 16 + r + ((lane >> 4) << 3);
        float v = acc[i][j][r] + bv;
        if (mode == 2) {
          outF[(size_t)gm * N + gn] = v;
        } else if (mode == 0) {
          int b = gm >> 11, s = gm & 2047;
          int h = gn >> 6,  dd = gn & 63;
          out0[(((size_t)(b * H_ + h)) * SQ_ + s) * HD_ + dd] = f2bf(v);
        } else {
          int b = gm >> 11, s = gm & 2047;
          if (gn < D_) {
            int h = gn >> 6, dd = gn & 63;
            out0[(((size_t)(b * H_ + h)) * SKV_ + s) * HD_ + dd] = f2bf(v);
          } else {
            int n2 = gn - D_;
            int h = n2 >> 6, dd = n2 & 63;
            out1[(((size_t)(b * H_ + h)) * HD_ + dd) * SKV_ + s] = f2bf(v);
          }
        }
      }
    }
  }
}

// ---------------- flash attention ----------------
// grid: (SQ/128, H, B); 8 waves/block; each wave owns a 16-row Q tile.
__global__ __launch_bounds__(256) void attn_kernel(
    const unsigned short* __restrict__ Q,    // [B,H,SQ,HD]  bf16
    const unsigned short* __restrict__ Kb,   // [B,H,SKV,HD] bf16
    const unsigned short* __restrict__ Vt,   // [B,H,HD,SKV] bf16
    const unsigned char*  __restrict__ mask, // [B,SKV] bool
    unsigned short* __restrict__ attnOut) {  // [B,SQ,D] bf16
  __shared__ __align__(16) unsigned short Pl[8][16 * 40];

  const int tid  = threadIdx.x;
  const int lane = tid & 31;
  const int w    = tid >> 5;
  const int b    = blockIdx.z;
  const int h    = blockIdx.y;
  const int m0   = blockIdx.x * 128 + w * 16;

  const unsigned short* Qh = Q  + (size_t)(b * H_ + h) * SQ_  * HD_;
  const unsigned short* Kh = Kb + (size_t)(b * H_ + h) * SKV_ * HD_;
  const unsigned short* Vh = Vt + (size_t)(b * H_ + h) * HD_  * SKV_;
  const unsigned char*  mb = mask + (size_t)b * SKV_;

  bf16x16 q0 = load_frag(Qh, m0, HD_, 0,  lane);
  bf16x16 q1 = load_frag(Qh, m0, HD_, 32, lane);

  f32x8 o[4] = {};
  float mrun[8], lrun[8];
#pragma unroll
  for (int r = 0; r < 8; ++r) { mrun[r] = -__builtin_inff(); lrun[r] = 0.0f; }

  const float scale = 0.125f; // 1/sqrt(64)

  for (int j0 = 0; j0 < SKV_; j0 += 32) {
    bf16x16 k00 = load_frag(Kh, j0,      HD_, 0,  lane);
    bf16x16 k01 = load_frag(Kh, j0,      HD_, 32, lane);
    bf16x16 k10 = load_frag(Kh, j0 + 16, HD_, 0,  lane);
    bf16x16 k11 = load_frag(Kh, j0 + 16, HD_, 32, lane);

    f32x8 s0 = {}, s1 = {};
    s0 = __builtin_amdgcn_wmma_f32_16x16x32_bf16(false, q0, false, k00, (short)0, s0, false, false);
    s0 = __builtin_amdgcn_wmma_f32_16x16x32_bf16(false, q1, false, k01, (short)0, s0, false, false);
    s1 = __builtin_amdgcn_wmma_f32_16x16x32_bf16(false, q0, false, k10, (short)0, s1, false, false);
    s1 = __builtin_amdgcn_wmma_f32_16x16x32_bf16(false, q1, false, k11, (short)0, s1, false, false);

    bool msk0 = mb[j0 +      (lane & 15)] != 0;
    bool msk1 = mb[j0 + 16 + (lane & 15)] != 0;

    float p0[8], p1[8];
#pragma unroll
    for (int r = 0; r < 8; ++r) {
      float v0 = msk0 ? -__builtin_inff() : s0[r] * scale;
      float v1 = msk1 ? -__builtin_inff() : s1[r] * scale;
      // row-wise reduce: a row lives in one 16-lane half; xor 1,2,4,8 stays in-half
      float rm = fmaxf(v0, v1);
      rm = fmaxf(rm, __shfl_xor(rm, 1, 32));
      rm = fmaxf(rm, __shfl_xor(rm, 2, 32));
      rm = fmaxf(rm, __shfl_xor(rm, 4, 32));
      rm = fmaxf(rm, __shfl_xor(rm, 8, 32));
      float mnew  = fmaxf(mrun[r], rm);
      float alpha = __expf(mrun[r] - mnew);
      p0[r] = __expf(v0 - mnew);
      p1[r] = __expf(v1 - mnew);
      float rs = p0[r] + p1[r];
      rs += __shfl_xor(rs, 1, 32);
      rs += __shfl_xor(rs, 2, 32);
      rs += __shfl_xor(rs, 4, 32);
      rs += __shfl_xor(rs, 8, 32);
      lrun[r] = lrun[r] * alpha + rs;
      mrun[r] = mnew;
#pragma unroll
      for (int t = 0; t < 4; ++t) o[t][r] *= alpha;
    }

    // round-trip P (C layout f32) through per-wave LDS to A-layout bf16
#pragma unroll
    for (int r = 0; r < 8; ++r) {
      int row = r + ((lane >> 4) << 3);
      Pl[w][row * 40 +      (lane & 15)] = f2bf(p0[r]);
      Pl[w][row * 40 + 16 + (lane & 15)] = f2bf(p1[r]);
    }
    asm volatile("s_wait_dscnt 0" ::: "memory"); // same-wave LDS RAW fence

    bf16x16 pa = load_frag(&Pl[w][0], 0, 40, 0, lane);
#pragma unroll
    for (int t = 0; t < 4; ++t) {
      bf16x16 vb = load_frag(Vh, t * 16, SKV_, j0, lane);
      o[t] = __builtin_amdgcn_wmma_f32_16x16x32_bf16(
          false, pa, false, vb, (short)0, o[t], false, false);
    }
  }

#pragma unroll
  for (int r = 0; r < 8; ++r) {
    float inv = 1.0f / lrun[r];
    int s = m0 + r + ((lane >> 4) << 3);
#pragma unroll
    for (int t = 0; t < 4; ++t) {
      int col = h * HD_ + t * 16 + (lane & 15);
      attnOut[((size_t)b * SQ_ + s) * D_ + col] = f2bf(o[t][r] * inv);
    }
  }
}

// ---------------- launch ----------------

extern "C" void kernel_launch(void* const* d_in, const int* in_sizes, int n_in,
                              void* d_out, int out_size, void* d_ws, size_t ws_size,
                              hipStream_t stream) {
  const float*         query   = (const float*)d_in[0];
  const float*         context = (const float*)d_in[1];
  const unsigned char* mask    = (const unsigned char*)d_in[2];
  const float*         Wq      = (const float*)d_in[3];
  const float*         bq      = (const float*)d_in[4];
  const float*         Wkv     = (const float*)d_in[5];
  const float*         bkv     = (const float*)d_in[6];
  const float*         Wout    = (const float*)d_in[7];
  const float*         bout    = (const float*)d_in[8];
  (void)in_sizes; (void)n_in; (void)out_size; (void)ws_size;

  char* ws = (char*)d_ws;
  size_t off = 0;
  auto carve = [&](size_t bytes) {
    char* p = ws + off;
    off += (bytes + 255) & ~(size_t)255;
    return p;
  };
  unsigned short* qbf  = (unsigned short*)carve((size_t)B_ * SQ_  * D_ * 2);
  unsigned short* cbf  = (unsigned short*)carve((size_t)B_ * SKV_ * D_ * 2);
  unsigned short* wqT  = (unsigned short*)carve((size_t)D_ * D_ * 2);
  unsigned short* wkvT = (unsigned short*)carve((size_t)2 * D_ * D_ * 2);
  unsigned short* woT  = (unsigned short*)carve((size_t)D_ * D_ * 2);
  unsigned short* Qb   = (unsigned short*)carve((size_t)B_ * H_ * SQ_  * HD_ * 2);
  unsigned short* Kbuf = (unsigned short*)carve((size_t)B_ * H_ * SKV_ * HD_ * 2);
  unsigned short* Vt   = (unsigned short*)carve((size_t)B_ * H_ * HD_ * SKV_ * 2);
  unsigned short* attn = (unsigned short*)carve((size_t)B_ * SQ_ * D_ * 2);

  cvt_bf16_kernel<<<1024, 256, 0, stream>>>(query,   qbf, B_ * SQ_  * D_);
  cvt_bf16_kernel<<<1024, 256, 0, stream>>>(context, cbf, B_ * SKV_ * D_);
  transpose_cvt_kernel<<<1024, 256, 0, stream>>>(Wq,   wqT,  D_, D_);
  transpose_cvt_kernel<<<1024, 256, 0, stream>>>(Wkv,  wkvT, D_, 2 * D_);
  transpose_cvt_kernel<<<1024, 256, 0, stream>>>(Wout, woT,  D_, D_);

  dim3 blk(256);
  // Q = query @ Wq + bq  -> [B,H,SQ,HD]
  gemm_bf16_kernel<<<dim3(D_ / 128, (B_ * SQ_) / 128), blk, 0, stream>>>(
      qbf, wqT, bq, B_ * SQ_, D_, D_, 0, Qb, nullptr, nullptr);
  // KV = context @ Wkv + bkv -> K [B,H,SKV,HD], V^T [B,H,HD,SKV]
  gemm_bf16_kernel<<<dim3((2 * D_) / 128, (B_ * SKV_) / 128), blk, 0, stream>>>(
      cbf, wkvT, bkv, B_ * SKV_, 2 * D_, D_, 1, Kbuf, Vt, nullptr);
  // flash attention -> attn [B,SQ,D] bf16
  attn_kernel<<<dim3(SQ_ / 128, H_, B_), blk, 0, stream>>>(Qb, Kbuf, Vt, mask, attn);
  // out = attn @ Wout + bout -> f32
  gemm_bf16_kernel<<<dim3(D_ / 128, (B_ * SQ_) / 128), blk, 0, stream>>>(
      attn, woT, bout, B_ * SQ_, D_, D_, 2, nullptr, nullptr, (float*)d_out);
}
